// SingleMambaBlockimage_25649544692420
// MI455X (gfx1250) — matrix operand order repaired and checked by reading, not compiled, not run on hardware
//
#include <hip/hip_runtime.h>
#include <hip/hip_bf16.h>
#include <stdint.h>

#define DIMD 768
#define BSZN 4
#define SEQL 2048
#define DSN  8
#define DRN  48
#define DCN  4
#define TOK  (BSZN * SEQL)   // 8192 tokens
#define CHK  16              // scan chunks
#define LCH  (SEQL / CHK)    // 128 steps per chunk
#define EPSF 1e-5f

typedef __attribute__((ext_vector_type(16))) __bf16       v16bf;
typedef __attribute__((ext_vector_type(8)))  float        v8f;
typedef __attribute__((ext_vector_type(4)))  int          v4i_;
typedef __attribute__((ext_vector_type(4)))  unsigned int v4u_;
typedef __attribute__((ext_vector_type(8)))  int          v8i_;

// ---- gfx1250 async global->LDS path (compile-time probed; confirmed present) ----
#if defined(__has_builtin)
#  if __has_builtin(__builtin_amdgcn_global_load_async_to_lds_b128) && \
      __has_builtin(__builtin_amdgcn_s_wait_asynccnt)
#    define HAVE_ASYNC_LDS 1
#  endif
#endif
#ifndef HAVE_ASYNC_LDS
#  define HAVE_ASYNC_LDS 0
#endif

// ---- gfx1250 Tensor Data Mover (compile-time probed; 6-arg clang-23 form) ----
#if defined(__has_builtin)
#  if __has_builtin(__builtin_amdgcn_tensor_load_to_lds) && \
      __has_builtin(__builtin_amdgcn_s_wait_tensorcnt) && HAVE_ASYNC_LDS
#    define HAVE_TDM 1
#  endif
#endif
#ifndef HAVE_TDM
#  define HAVE_TDM 0
#endif

#if HAVE_ASYNC_LDS
typedef __attribute__((address_space(1))) v4i_ g_v4i;   // global
typedef __attribute__((address_space(3))) v4i_ l_v4i;   // LDS
#endif

// ---------- helpers ----------
__device__ __forceinline__ unsigned short f2bf(float f) {
  union { float f; unsigned int u; } x; x.f = f;
  unsigned int u = x.u;
  u += 0x7FFFu + ((u >> 16) & 1u);   // round-to-nearest-even
  return (unsigned short)(u >> 16);
}
__device__ __forceinline__ float sigmoidf_(float x) { return 1.0f / (1.0f + __expf(-x)); }
__device__ __forceinline__ float siluf_(float x)    { return x * sigmoidf_(x); }
__device__ __forceinline__ float softplusf_(float x){ return (x > 20.0f) ? x : __logf(1.0f + __expf(x)); }

#if HAVE_TDM
// TDM descriptor load of a [rows x 64] bf16 tile (row stride K elements) into LDS with
// hardware row padding: 16B pad after every 128B -> lands in ushort[rows][72] layout.
// ISA 8.3/8.4 D# packing. Issued once per wave; tracked with TENSORcnt.
__device__ __forceinline__ void tdm_load_tile(const unsigned short* gptr, unsigned lds_off,
                                              int K, int rows) {
  unsigned long long ga = (unsigned long long)(uintptr_t)gptr;
  v4u_ g0;
  g0[0] = 1u;                                             // count=1 (valid), user mode
  g0[1] = lds_off;                                        // lds_addr (bytes)
  g0[2] = (unsigned)(ga & 0xFFFFFFFFu);                   // global_addr[31:0]
  g0[3] = (unsigned)((ga >> 32) & 0x01FFFFFFu)            // global_addr[56:32]
        | (2u << 30);                                     // type=2 ("image")
  unsigned td0 = (unsigned)K;                             // tensor_dim0 (elements)
  unsigned td1 = (unsigned)rows;                          // tensor_dim1
  v8i_ g1;
  g1[0] = (int)((1u << 16)                                // data_size = 2 bytes
              | (1u << 20)                                // pad_enable
              | (4u << 22)                                // pad_interval: 32 DWORDs (128B)
              | (3u << 25));                              // pad_amount: 4 DWORDs (16B)
  g1[1] = (int)((td0 & 0xFFFFu) << 16);                   // atomic_addr=0 | tensor_dim0[15:0]
  g1[2] = (int)(((td0 >> 16) & 0xFFFFu) | ((td1 & 0xFFFFu) << 16));
  g1[3] = (int)(((td1 >> 16) & 0xFFFFu) | (64u << 16));   // tile_dim0 = 64
  g1[4] = (int)(td1 & 0xFFFFu);                           // tile_dim1 = rows; tile_dim2 = 0
  g1[5] = (int)(unsigned)K;                               // tensor_dim0_stride[31:0] = K
  g1[6] = 0;                                              // stride[47:32]=0; dim1_stride lo=0
  g1[7] = 0;
  v4i_ z4 = {0, 0, 0, 0};                                 // groups 2/3 unused (2-D tensor)
  v8i_ z8 = {0, 0, 0, 0, 0, 0, 0, 0};
  __builtin_amdgcn_tensor_load_to_lds(g0, g1, z4, z4, z8, 0);
}
#endif

// ---------- LayerNorm -> bf16 (one block per token) ----------
__global__ __launch_bounds__(256)
void ln_kernel(const float* __restrict__ x, const float* __restrict__ g,
               const float* __restrict__ b, unsigned short* __restrict__ xn) {
  int t = blockIdx.x;
  const float* xr = x + (size_t)t * DIMD;
  __shared__ float red[256];
  float s = 0.0f;
  for (int i = threadIdx.x; i < DIMD; i += 256) s += xr[i];
  red[threadIdx.x] = s; __syncthreads();
  for (int o = 128; o > 0; o >>= 1) { if (threadIdx.x < o) red[threadIdx.x] += red[threadIdx.x + o]; __syncthreads(); }
  float mu = red[0] * (1.0f / DIMD); __syncthreads();
  float v = 0.0f;
  for (int i = threadIdx.x; i < DIMD; i += 256) { float d = xr[i] - mu; v += d * d; }
  red[threadIdx.x] = v; __syncthreads();
  for (int o = 128; o > 0; o >>= 1) { if (threadIdx.x < o) red[threadIdx.x] += red[threadIdx.x + o]; __syncthreads(); }
  float rstd = rsqrtf(red[0] * (1.0f / DIMD) + EPSF);
  for (int i = threadIdx.x; i < DIMD; i += 256)
    xn[(size_t)t * DIMD + i] = f2bf((xr[i] - mu) * rstd * g[i] + b[i]);
}

// ---------- fp32 (N,K) weight -> bf16 (N,Kp) zero-padded ----------
__global__ __launch_bounds__(256)
void wcvt_kernel(const float* __restrict__ src, unsigned short* __restrict__ dst,
                 int N, int K, int Kp) {
  int i = blockIdx.x * 256 + threadIdx.x;
  if (i >= N * Kp) return;
  int n = i / Kp, k = i % Kp;
  dst[i] = (k < K) ? f2bf(src[(size_t)n * K + k]) : (unsigned short)0;
}

// ---------- bf16 WMMA GEMM:  C(MxN) = A(MxK) * W(NxK)^T ----------
// Tile 128(M) x 64(N), Ktile 64; 256 threads = 8 waves (4x2); 8 WMMAs/wave/K-tile.
// Double-buffered LDS; A tile fed by TDM (TENSORcnt), W tile by async DMA (ASYNCcnt).
template <bool HAS_RES, bool BF16OUT>
__global__ __launch_bounds__(256)
void gemm_bf16_kernel(const unsigned short* __restrict__ A,
                      const unsigned short* __restrict__ W,
                      float* __restrict__ C,
                      const float* __restrict__ resid,
                      unsigned short* __restrict__ Cbf,
                      int M, int N, int K) {
  __shared__ unsigned short As[2][128][72];   // 64 used + 16B row pad (conflict-free)
  __shared__ unsigned short Ws[2][64][72];
  int tid  = threadIdx.x;
  int lane = tid & 31, wave = tid >> 5;
  int wm = wave >> 1, wn = wave & 1;           // waves: 4 (M) x 2 (N)
  int half = lane >> 4, l16 = lane & 15;
  int rowBase = blockIdx.y * 128;
  int colBase = blockIdx.x * 64;

  v8f acc[2][2] = {};

  auto issueA = [&](int buf, int k0) {
#if HAVE_TDM
    if (wave == 0) {
      unsigned lds_off = (unsigned)(uintptr_t)(l_v4i*)&As[buf][0][0];
      tdm_load_tile(&A[(size_t)rowBase * K + k0], lds_off, K, 128);
    }
#elif HAVE_ASYNC_LDS
    #pragma unroll
    for (int i = 0; i < 4; ++i) {              // A tile: 128x64 bf16 = 1024 x b128
      int ii = tid + i * 256;
      int r = ii >> 3, c = (ii & 7) << 3;
      __builtin_amdgcn_global_load_async_to_lds_b128(
          (g_v4i*)&A[(size_t)(rowBase + r) * K + k0 + c], (l_v4i*)&As[buf][r][c], 0, 0);
    }
#else
    #pragma unroll
    for (int i = 0; i < 4; ++i) {
      int ii = tid + i * 256;
      int r = ii >> 3, c = (ii & 7) << 3;
      *(uint4*)&As[buf][r][c] = *(const uint4*)&A[(size_t)(rowBase + r) * K + k0 + c];
    }
    __builtin_prefetch(&A[(size_t)(rowBase + (tid & 127)) * K + k0 + 64], 0, 0);
#endif
  };

  auto issueW = [&](int buf, int k0) {
#if HAVE_ASYNC_LDS
    #pragma unroll
    for (int i = 0; i < 2; ++i) {              // W tile: 64x64 bf16 = 512 x b128
      int ii = tid + i * 256;
      int r = ii >> 3, c = (ii & 7) << 3;
      __builtin_amdgcn_global_load_async_to_lds_b128(
          (g_v4i*)&W[(size_t)(colBase + r) * K + k0 + c], (l_v4i*)&Ws[buf][r][c], 0, 0);
    }
#else
    #pragma unroll
    for (int i = 0; i < 2; ++i) {
      int ii = tid + i * 256;
      int r = ii >> 3, c = (ii & 7) << 3;
      *(uint4*)&Ws[buf][r][c] = *(const uint4*)&W[(size_t)(colBase + r) * K + k0 + c];
    }
#endif
  };

  int T = K / 64;
  issueA(0, 0);
  issueW(0, 0);
  for (int t = 0; t < T; ++t) {
    int cur = t & 1;
    if (t + 1 < T) { issueA(cur ^ 1, (t + 1) * 64); issueW(cur ^ 1, (t + 1) * 64); }
#if HAVE_TDM
    if (wave == 0) {                                        // A tile via TENSORcnt
      if (t + 1 < T) __builtin_amdgcn_s_wait_tensorcnt(1);
      else           __builtin_amdgcn_s_wait_tensorcnt(0);
    }
    if (t + 1 < T) __builtin_amdgcn_s_wait_asynccnt(2);     // W tile via ASYNCcnt
    else           __builtin_amdgcn_s_wait_asynccnt(0);
#elif HAVE_ASYNC_LDS
    if (t + 1 < T) __builtin_amdgcn_s_wait_asynccnt(6);     // oldest 6 (current tile) done
    else           __builtin_amdgcn_s_wait_asynccnt(0);
#endif
    __syncthreads();

    #pragma unroll
    for (int kk = 0; kk < 64; kk += 32) {
      union { v16bf v; unsigned int u[8]; } af[2], bf2[2];
      #pragma unroll
      for (int mi = 0; mi < 2; ++mi) {
        int row = wm * 32 + mi * 16 + l16;
        #pragma unroll
        for (int v = 0; v < 8; ++v) {
          // ISA 16-bit A 16x32 layout: lanes0-15 K={0..7,16..23}, lanes16-31 K={8..15,24..31}
          int kf = (v < 4) ? (half * 8 + v * 2) : (16 + half * 8 + (v - 4) * 2);
          af[mi].u[v] = *(const unsigned int*)&As[cur][row][kk + kf];
        }
      }
      #pragma unroll
      for (int ni = 0; ni < 2; ++ni) {
        int col = wn * 32 + ni * 16 + l16;
        #pragma unroll
        for (int v = 0; v < 8; ++v) {
          // B 32x16: lanes0-15 K={0..7,16..23}, lanes16-31 K={8..15,24..31}; N = lane&15
          int kf = (v < 4) ? (half * 8 + v * 2) : (16 + half * 8 + (v - 4) * 2);
          bf2[ni].u[v] = *(const unsigned int*)&Ws[cur][col][kk + kf];
        }
      }
      #pragma unroll
      for (int mi = 0; mi < 2; ++mi)
        #pragma unroll
        for (int ni = 0; ni < 2; ++ni)
          acc[mi][ni] = __builtin_amdgcn_wmma_f32_16x16x32_bf16(
              false, af[mi].v, false, bf2[ni].v, (short)0, acc[mi][ni], false, false);
    }
    __syncthreads();   // all waves done reading `cur` before it is refilled at t+2
  }

  #pragma unroll
  for (int mi = 0; mi < 2; ++mi)
    #pragma unroll
    for (int ni = 0; ni < 2; ++ni) {
      int col = colBase + wn * 32 + ni * 16 + l16;
      #pragma unroll
      for (int v = 0; v < 8; ++v) {
        int row = rowBase + wm * 32 + mi * 16 + half * 8 + v;   // C/D f32 layout
        float val = acc[mi][ni][v];
        if (HAS_RES) val += resid[(size_t)row * N + col];
        C[(size_t)row * N + col] = val;
        if (BF16OUT) Cbf[(size_t)row * N + col] = f2bf(val);
      }
    }
}

// ---------- causal depthwise conv + SiLU (backward dir written in reversed order) ----------
__global__ __launch_bounds__(256)
void conv_silu_kernel(const float* __restrict__ xz,        // (TOK, 2*DIMD), xf = cols [0,768)
                      const float* __restrict__ w, const float* __restrict__ bias,
                      float* __restrict__ xs_f, unsigned short* __restrict__ xs_bf,
                      int reverse) {
  int idx = blockIdx.x * 256 + threadIdx.x;
  if (idx >= TOK * DIMD) return;
  int d = idx % DIMD;
  int t = idx / DIMD;
  int b = t / SEQL, j = t % SEQL;                 // j = position in (maybe reversed) sequence
  float acc = bias[d];
  #pragma unroll
  for (int k = 0; k < DCN; ++k) {
    int i = j - (DCN - 1) + k;
    if (i >= 0) {
      int l = reverse ? (SEQL - 1 - i) : i;       // original index
      acc += w[d * DCN + k] * xz[((size_t)(b * SEQL + l)) * (2 * DIMD) + d];
    }
  }
  float s = siluf_(acc);
  xs_f[idx]  = s;
  xs_bf[idx] = f2bf(s);
}

// ======== chunked selective scan (3 passes; affine recurrence h' = dA*h + dBx) ========
// Storage layout for chunk summaries: [(b*DIMD+d)*CHK + c]*DSN + n

__global__ __launch_bounds__(256)
void scan_chunk_kernel(const float* __restrict__ delta_pre, const float* __restrict__ dt_bias,
                       const float* __restrict__ xdbl, const float* __restrict__ xs,
                       const float* __restrict__ A_log,
                       float* __restrict__ Pc, float* __restrict__ Hc) {
  int idx = blockIdx.x * 256 + threadIdx.x;
  if (idx >= BSZN * DIMD * DSN * CHK) return;
  int n = idx & (DSN - 1);
  int rest = idx >> 3;
  int c  = rest % CHK;
  int bd = rest / CHK;
  int b = bd / DIMD, d = bd % DIMD;

  float An = -__expf(A_log[d * DSN + n]);
  float db = dt_bias[d];
  float P = 1.0f, H = 0.0f;
  for (int s = 0; s < LCH; ++s) {
    size_t t = (size_t)b * SEQL + c * LCH + s;
    float delta = softplusf_(delta_pre[t * DIMD + d] + db);
    float dA = __expf(delta * An);
    P *= dA;
    H = dA * H + (delta * xs[t * DIMD + d]) * xdbl[t * 64 + 48 + n];
  }
  size_t o = ((size_t)bd * CHK + c) * DSN + n;
  Pc[o] = P; Hc[o] = H;
}

__global__ __launch_bounds__(256)
void scan_combine_kernel(const float* __restrict__ Pc, const float* __restrict__ Hc,
                         float* __restrict__ hin) {
  int idx = blockIdx.x * 256 + threadIdx.x;
  if (idx >= BSZN * DIMD * DSN) return;
  int n = idx & (DSN - 1);
  int bd = idx >> 3;
  float h = 0.0f;
  for (int c = 0; c < CHK; ++c) {
    size_t o = ((size_t)bd * CHK + c) * DSN + n;
    hin[o] = h;
    h = Pc[o] * h + Hc[o];
  }
}

__global__ __launch_bounds__(256)
void scan_emit_kernel(const float* __restrict__ delta_pre, const float* __restrict__ dt_bias,
                      const float* __restrict__ xdbl, const float* __restrict__ xs,
                      const float* __restrict__ A_log, const float* __restrict__ Dp,
                      const float* __restrict__ xz,   // z = cols [768,1536), ORIGINAL order
                      const float* __restrict__ hin,
                      float* __restrict__ y, int reverse) {
  int idx = blockIdx.x * 256 + threadIdx.x;
  if (idx >= BSZN * DIMD * DSN * CHK) return;
  int n = idx & (DSN - 1);            // == lane&7: groups of 8 lanes share (b,d,c)
  int rest = idx >> 3;
  int c  = rest % CHK;
  int bd = rest / CHK;
  int b = bd / DIMD, d = bd % DIMD;

  float An = -__expf(A_log[d * DSN + n]);
  float Dv = Dp[d], db = dt_bias[d];
  float h = hin[((size_t)bd * CHK + c) * DSN + n];

  for (int s = 0; s < LCH; ++s) {
    int j = c * LCH + s;
    size_t t = (size_t)b * SEQL + j;
    float delta = softplusf_(delta_pre[t * DIMD + d] + db);
    float xv    = xs[t * DIMD + d];
    float dA    = __expf(delta * An);
    h = dA * h + (delta * xv) * xdbl[t * 64 + 48 + n];
    float yv = h * xdbl[t * 64 + 56 + n];
    yv += __shfl_xor(yv, 1, 32);
    yv += __shfl_xor(yv, 2, 32);
    yv += __shfl_xor(yv, 4, 32);
    if (n == 0) {
      int l = reverse ? (SEQL - 1 - j) : j;
      float z = xz[((size_t)b * SEQL + l) * (2 * DIMD) + DIMD + d];
      y[t * DIMD + d] = (yv + xv * Dv) * siluf_(z);
    }
  }
}

// ---------- combine yf + flipped yb -> bf16 ----------
__global__ __launch_bounds__(256)
void combine_kernel(const float* __restrict__ yf, const float* __restrict__ yb_rev,
                    unsigned short* __restrict__ yc) {
  int idx = blockIdx.x * 256 + threadIdx.x;
  if (idx >= TOK * DIMD) return;
  int d = idx % DIMD, t = idx / DIMD;
  int b = t / SEQL, l = t % SEQL;
  size_t rb = ((size_t)b * SEQL + (SEQL - 1 - l)) * DIMD + d;
  yc[idx] = f2bf(0.5f * (yf[idx] + yb_rev[rb]));
}

// =====================================================================
extern "C" void kernel_launch(void* const* d_in, const int* in_sizes, int n_in,
                              void* d_out, int out_size, void* d_ws, size_t ws_size,
                              hipStream_t stream) {
  (void)in_sizes; (void)n_in; (void)out_size; (void)ws_size;
  const float* x         = (const float*)d_in[0];
  const float* ln_g      = (const float*)d_in[1];
  const float* ln_b      = (const float*)d_in[2];
  const float* in_proj_w = (const float*)d_in[3];
  const float* conv_w    = (const float*)d_in[4];
  const float* conv_b    = (const float*)d_in[5];
  const float* x_proj_w  = (const float*)d_in[6];
  const float* dt_proj_w = (const float*)d_in[7];
  const float* dt_proj_b = (const float*)d_in[8];
  const float* A_log     = (const float*)d_in[9];
  const float* Dvec      = (const float*)d_in[10];
  const float* conv_w_b  = (const float*)d_in[11];
  const float* conv_b_b  = (const float*)d_in[12];
  const float* x_proj_wb = (const float*)d_in[13];
  const float* dt_proj_wb= (const float*)d_in[14];
  const float* dt_proj_bb= (const float*)d_in[15];
  const float* A_b_log   = (const float*)d_in[16];
  const float* D_b       = (const float*)d_in[17];
  const float* out_proj_w= (const float*)d_in[18];
  float* out = (float*)d_out;

  // ---- workspace bump allocator (256B aligned) ----
  char* ws = (char*)d_ws;
  size_t off = 0;
  auto alloc = [&](size_t bytes) -> char* {
    char* p = ws + off;
    off += (bytes + 255) & ~(size_t)255;
    return p;
  };
  unsigned short* xn_bf   = (unsigned short*)alloc((size_t)TOK * DIMD * 2);
  float*          xzbuf   = (float*)         alloc((size_t)TOK * 2 * DIMD * 4);
  float*          xsf_f   = (float*)         alloc((size_t)TOK * DIMD * 4);
  unsigned short* xsf_bf  = (unsigned short*)alloc((size_t)TOK * DIMD * 2);
  float*          xsb_f   = (float*)         alloc((size_t)TOK * DIMD * 4);
  unsigned short* xsb_bf  = (unsigned short*)alloc((size_t)TOK * DIMD * 2);
  float*          xdbl_f  = (float*)         alloc((size_t)TOK * 64 * 4);
  float*          xdbl_b  = (float*)         alloc((size_t)TOK * 64 * 4);
  unsigned short* dtp_f   = (unsigned short*)alloc((size_t)TOK * 64 * 2);
  unsigned short* dtp_b   = (unsigned short*)alloc((size_t)TOK * 64 * 2);
  float*          dpre_f  = (float*)         alloc((size_t)TOK * DIMD * 4);
  float*          dpre_b  = (float*)         alloc((size_t)TOK * DIMD * 4);
  float*          yf      = (float*)         alloc((size_t)TOK * DIMD * 4);
  float*          yb      = (float*)         alloc((size_t)TOK * DIMD * 4);
  unsigned short* yc      = (unsigned short*)alloc((size_t)TOK * DIMD * 2);
  size_t nsum = (size_t)BSZN * DIMD * DSN * CHK;     // 393216 chunk summaries
  float*          Pc      = (float*)         alloc(nsum * 4);
  float*          Hc      = (float*)         alloc(nsum * 4);
  float*          hin     = (float*)         alloc(nsum * 4);
  unsigned short* w_in    = (unsigned short*)alloc((size_t)2 * DIMD * DIMD * 2);
  unsigned short* w_xpf   = (unsigned short*)alloc((size_t)64 * DIMD * 2);
  unsigned short* w_xpb   = (unsigned short*)alloc((size_t)64 * DIMD * 2);
  unsigned short* w_dtf   = (unsigned short*)alloc((size_t)DIMD * 64 * 2);
  unsigned short* w_dtb   = (unsigned short*)alloc((size_t)DIMD * 64 * 2);
  unsigned short* w_out   = (unsigned short*)alloc((size_t)DIMD * DIMD * 2);

  dim3 blk256(256);
  int elemGrid  = (TOK * DIMD + 255) / 256;
  int chunkGrid = ((int)nsum + 255) / 256;                  // 1536 blocks
  int bdnGrid   = (BSZN * DIMD * DSN + 255) / 256;          // 96 blocks

  // 1) LayerNorm -> bf16
  ln_kernel<<<TOK, blk256, 0, stream>>>(x, ln_g, ln_b, xn_bf);

  // 2) weight conversions (dt_proj K padded 48 -> 64 with zeros)
  wcvt_kernel<<<(2 * DIMD * DIMD + 255) / 256, blk256, 0, stream>>>(in_proj_w,  w_in,  2 * DIMD, DIMD, DIMD);
  wcvt_kernel<<<(64 * DIMD + 255) / 256,       blk256, 0, stream>>>(x_proj_w,   w_xpf, 64,  DIMD, DIMD);
  wcvt_kernel<<<(64 * DIMD + 255) / 256,       blk256, 0, stream>>>(x_proj_wb,  w_xpb, 64,  DIMD, DIMD);
  wcvt_kernel<<<(DIMD * 64 + 255) / 256,       blk256, 0, stream>>>(dt_proj_w,  w_dtf, DIMD, DRN, 64);
  wcvt_kernel<<<(DIMD * 64 + 255) / 256,       blk256, 0, stream>>>(dt_proj_wb, w_dtb, DIMD, DRN, 64);
  wcvt_kernel<<<(DIMD * DIMD + 255) / 256,     blk256, 0, stream>>>(out_proj_w, w_out, DIMD, DIMD, DIMD);

  // 3) in_proj GEMM: xz = xn @ in_proj_w^T   (M=8192, N=1536, K=768)
  gemm_bf16_kernel<false, false><<<dim3((2 * DIMD) / 64, TOK / 128), blk256, 0, stream>>>(
      xn_bf, w_in, xzbuf, nullptr, nullptr, TOK, 2 * DIMD, DIMD);

  // 4) depthwise conv + SiLU (fwd; bwd materialized reversed)
  conv_silu_kernel<<<elemGrid, blk256, 0, stream>>>(xzbuf, conv_w,   conv_b,   xsf_f, xsf_bf, 0);
  conv_silu_kernel<<<elemGrid, blk256, 0, stream>>>(xzbuf, conv_w_b, conv_b_b, xsb_f, xsb_bf, 1);

  // 5) x_proj GEMMs (M=8192, N=64, K=768); epilogue emits bf16 copy feeding dt_proj
  gemm_bf16_kernel<false, true><<<dim3(1, TOK / 128), blk256, 0, stream>>>(
      xsf_bf, w_xpf, xdbl_f, nullptr, dtp_f, TOK, 64, DIMD);
  gemm_bf16_kernel<false, true><<<dim3(1, TOK / 128), blk256, 0, stream>>>(
      xsb_bf, w_xpb, xdbl_b, nullptr, dtp_b, TOK, 64, DIMD);

  // 6) dt_proj GEMMs: delta_pre = dt @ dt_proj_w^T  (M=8192, N=768, K=64)
  gemm_bf16_kernel<false, false><<<dim3(DIMD / 64, TOK / 128), blk256, 0, stream>>>(
      dtp_f, w_dtf, dpre_f, nullptr, nullptr, TOK, DIMD, 64);
  gemm_bf16_kernel<false, false><<<dim3(DIMD / 64, TOK / 128), blk256, 0, stream>>>(
      dtp_b, w_dtb, dpre_b, nullptr, nullptr, TOK, DIMD, 64);

  // 7) chunked selective scans (forward, then backward reusing summary buffers)
  scan_chunk_kernel  <<<chunkGrid, blk256, 0, stream>>>(dpre_f, dt_proj_b, xdbl_f, xsf_f, A_log, Pc, Hc);
  scan_combine_kernel<<<bdnGrid,   blk256, 0, stream>>>(Pc, Hc, hin);
  scan_emit_kernel   <<<chunkGrid, blk256, 0, stream>>>(dpre_f, dt_proj_b, xdbl_f, xsf_f, A_log,
                                                        Dvec, xzbuf, hin, yf, 0);
  scan_chunk_kernel  <<<chunkGrid, blk256, 0, stream>>>(dpre_b, dt_proj_bb, xdbl_b, xsb_f, A_b_log, Pc, Hc);
  scan_combine_kernel<<<bdnGrid,   blk256, 0, stream>>>(Pc, Hc, hin);
  scan_emit_kernel   <<<chunkGrid, blk256, 0, stream>>>(dpre_b, dt_proj_bb, xdbl_b, xsb_f, A_b_log,
                                                        D_b, xzbuf, hin, yb, 1);

  // 8) combine directions -> bf16
  combine_kernel<<<elemGrid, blk256, 0, stream>>>(yf, yb, yc);

  // 9) out_proj GEMM with fused residual: out = yc @ out_proj_w^T + x
  gemm_bf16_kernel<true, false><<<dim3(DIMD / 64, TOK / 128), blk256, 0, stream>>>(
      yc, w_out, out, x, nullptr, TOK, DIMD, DIMD);
}
